// SinsPlusSinkhornAttn_18897856102837
// MI455X (gfx1250) — compile-verified, matrix operand order checked by reading, not run on hardware
//
#include <hip/hip_runtime.h>

// ---------------------------------------------------------------------------
// Types / helpers for CDNA5 WMMA (gfx1250, wave32)
// ---------------------------------------------------------------------------
typedef __attribute__((ext_vector_type(16))) __bf16 v16bf;
typedef __attribute__((ext_vector_type(8)))  float  v8f;

union BV16 { uint4 q[2]; __bf16 h[16]; v16bf v; };

__device__ __forceinline__ v16bf ld16bf(const __bf16* p0, const __bf16* p1) {
  BV16 t;
  t.q[0] = *(const uint4*)p0;   // halves j=0..7
  t.q[1] = *(const uint4*)p1;   // halves j=8..15
  return t.v;
}

__device__ __forceinline__ void ld8f(const float* p, float* f) {
  float4 a = *(const float4*)p;
  float4 b = *(const float4*)(p + 4);
  f[0]=a.x; f[1]=a.y; f[2]=a.z; f[3]=a.w;
  f[4]=b.x; f[5]=b.y; f[6]=b.z; f[7]=b.w;
}

__device__ __forceinline__ v16bf cvt16(const float* f) {
  BV16 t;
#pragma unroll
  for (int j = 0; j < 16; ++j) t.h[j] = (__bf16)f[j];   // native v_cvt (packed)
  return t.v;
}

__device__ __forceinline__ v8f vzero8() {
  v8f v = {0.f,0.f,0.f,0.f,0.f,0.f,0.f,0.f};
  return v;
}

__device__ __forceinline__ v8f wmma_bf16(v16bf a, v16bf b, v8f c) {
  // D = A(16x32 bf16) x B(32x16 bf16) + C(16x16 f32)
  return __builtin_amdgcn_wmma_f32_16x16x32_bf16(
      false, a, false, b, (short)0, c, false, false);
}

// Problem sizes (fixed by the reference)
#define NB 8       // batch
#define NP 4096    // params
#define NT 128     // query tokens
#define ND 256     // model dim == D_EMBED

// ---------------------------------------------------------------------------
// K0a: q = query_tokens @ Wq  (tiny, fp32 VALU, bf16 out)
// ---------------------------------------------------------------------------
__global__ void __launch_bounds__(256)
k_qproj(const float* __restrict__ qt, const float* __restrict__ Wq,
        __bf16* __restrict__ qb) {
  int t = blockIdx.x, e = threadIdx.x;
  float acc = 0.f;
  for (int d = 0; d < ND; ++d) acc += qt[t*ND + d] * Wq[d*ND + e];
  qb[t*ND + e] = (__bf16)acc;
}

// ---------------------------------------------------------------------------
// K0b: pack Wk / Wv into WMMA B-tile layout (per-lane-contiguous 16 halves).
// Tile(kt,nt): lane L covers column n = nt*16+(L&15), K half selected by L>=16.
// ---------------------------------------------------------------------------
__global__ void __launch_bounds__(256)
k_pack(const float* __restrict__ Wk, const float* __restrict__ Wv,
       __bf16* __restrict__ wkp, __bf16* __restrict__ wvp) {
  int gid  = blockIdx.x * 256 + threadIdx.x;   // 0..8191
  int wsel = gid >> 12;
  int rem  = gid & 4095;
  int tile = rem >> 5;        // 0..127  (kt*16+nt)
  int lane = rem & 31;
  int kt = tile >> 4, nt = tile & 15;
  int lo = lane & 15, dl = (lane < 16) ? 0 : 8;
  const float* src = wsel ? Wv : Wk;
  __bf16* dst = wsel ? wvp : wkp;
  int n = nt*16 + lo;
#pragma unroll
  for (int j = 0; j < 16; ++j) {
    int k = kt*32 + dl + ((j < 8) ? j : j + 8);
    dst[tile*512 + lane*16 + j] = (__bf16)src[k*ND + n];
  }
}

__global__ void __launch_bounds__(256) k_zero(float* v) {
  v[blockIdx.x * 256 + threadIdx.x] = 0.f;
}

// ---------------------------------------------------------------------------
// K1: embeds[b,p,d] = sum_e enc(x[b,p])[e] * proj[p,d,e]   (bf16 out)
// One block per p.  A = enc (M=8 batch rows of 16, rest zeroed), K = e (256),
// N = d (256, 2 tiles per wave).  proj read exactly once -> 1 GiB stream.
// ---------------------------------------------------------------------------
__global__ void __launch_bounds__(256)
k_embed(const float* __restrict__ x, const float* __restrict__ proj,
        __bf16* __restrict__ embeds) {
  const int p = blockIdx.x;
  const int tid = threadIdx.x;
  __shared__ __align__(16) __bf16 encA[16 * ND];   // 8 KB

  { // phase 1: sinusoidal encodings for all 8 batch rows, rows 8..15 = 0
    int e = tid;
    float kk = (float)((e < 128) ? e : e - 128);
    float basis = __expf(kk * (-9.210340371976184f / 128.0f)); // 10000^{-k/128}
#pragma unroll
    for (int b = 0; b < 8; ++b) {
      float xv  = x[b * NP + p];
      float ang = xv * basis;
      float val = (e < 128) ? __cosf(ang) : __sinf(ang);
      encA[b * ND + e]       = (__bf16)val;
      encA[(b + 8) * ND + e] = (__bf16)0.0f;
    }
  }
  __syncthreads();

  const int w = tid >> 5, lane = tid & 31;
  const int lo = lane & 15;
  const int dl = (lane < 16) ? 0 : 8;
  const int d0 = w * 32;                        // this wave: d tiles d0, d0+16
  v8f acc0 = vzero8(), acc1 = vzero8();
  const float* rowB0 = proj + (size_t)p * (ND*ND) + (size_t)(d0 + lo) * ND;
  const float* rowB1 = rowB0 + 16 * ND;

#pragma unroll
  for (int kt = 0; kt < 8; ++kt) {
    int eb = kt * 32 + dl;
    v16bf A = ld16bf(&encA[lo * ND + eb], &encA[lo * ND + eb + 16]);
    float f0[16], f1[16];
    ld8f(rowB0 + eb,      f0);     ld8f(rowB0 + eb + 16, f0 + 8);
    ld8f(rowB1 + eb,      f1);     ld8f(rowB1 + eb + 16, f1 + 8);
    acc0 = wmma_bf16(A, cvt16(f0), acc0);
    acc1 = wmma_bf16(A, cvt16(f1), acc1);
  }
  if (lane < 16) {
#pragma unroll
    for (int r = 0; r < 8; ++r) {   // rows M=0..7 == batch b
      size_t m = (size_t)r * NP + p;
      embeds[m * ND + d0 + lo]      = (__bf16)acc0[r];
      embeds[m * ND + d0 + 16 + lo] = (__bf16)acc1[r];
    }
  }
}

// ---------------------------------------------------------------------------
// K2: GEMM [32768,256] x [256,256]:  k = embeds@Wk (row store),
//                                    v = embeds@Wv (stored transposed [b][e][p])
// ---------------------------------------------------------------------------
__global__ void __launch_bounds__(256)
k_kv(const __bf16* __restrict__ Ebf, const __bf16* __restrict__ Wpack,
     __bf16* __restrict__ outb, int transposed) {
  const int tid = threadIdx.x, w = tid >> 5, lane = tid & 31;
  const int lo = lane & 15, dl = (lane < 16) ? 0 : 8;
  const int m0 = (blockIdx.x * 8 + w) * 16;
  v8f acc[16];
#pragma unroll
  for (int i = 0; i < 16; ++i) acc[i] = vzero8();
  const __bf16* arow = Ebf + (size_t)(m0 + lo) * ND;

  for (int kt = 0; kt < 8; ++kt) {
    int eb = kt * 32 + dl;
    v16bf A = ld16bf(arow + eb, arow + eb + 16);
#pragma unroll
    for (int nt = 0; nt < 16; ++nt) {
      const __bf16* bp = Wpack + (size_t)((kt*16 + nt) * 512) + lane*16;
      v16bf B = ld16bf(bp, bp + 8);
      acc[nt] = wmma_bf16(A, B, acc[nt]);
    }
  }
  int mofs = (lane < 16) ? 0 : 8;
  if (!transposed) {
#pragma unroll
    for (int nt = 0; nt < 16; ++nt) {
      int e = nt*16 + lo;
#pragma unroll
      for (int r = 0; r < 8; ++r) {
        int m = m0 + r + mofs;
        outb[(size_t)m * ND + e] = (__bf16)acc[nt][r];
      }
    }
  } else {
#pragma unroll
    for (int nt = 0; nt < 16; ++nt) {
      int e = nt*16 + lo;
#pragma unroll
      for (int r = 0; r < 8; ++r) {
        int m = m0 + r + mofs;
        int b = m >> 12, p = m & (NP - 1);
        outb[(size_t)b * (ND*NP) + (size_t)e * NP + p] = (__bf16)acc[nt][r];
      }
    }
  }
}

// ---------------------------------------------------------------------------
// K3a: logits[b,t,n] = sum_d q[t,d]*k[b,n,d]   (f32 out, SINKHORN_REG = 1)
// ---------------------------------------------------------------------------
__global__ void __launch_bounds__(256)
k_logits(const __bf16* __restrict__ qb, const __bf16* __restrict__ kb,
         float* __restrict__ logits) {
  int bx = blockIdx.x;
  int b = bx >> 5, g = bx & 31;
  int tid = threadIdx.x, w = tid >> 5, lane = tid & 31;
  int lo = lane & 15, dl = (lane < 16) ? 0 : 8;
  int n0 = (g * 8 + w) * 16;
  v8f acc[8];
#pragma unroll
  for (int i = 0; i < 8; ++i) acc[i] = vzero8();
  const __bf16* brow = kb + (size_t)(b * NP + n0 + lo) * ND;

  for (int kt = 0; kt < 8; ++kt) {
    int eb = kt * 32 + dl;
    v16bf B = ld16bf(brow + eb, brow + eb + 16);
#pragma unroll
    for (int tt = 0; tt < 8; ++tt) {
      const __bf16* ar = qb + (size_t)(tt*16 + lo) * ND;
      v16bf A = ld16bf(ar + eb, ar + eb + 16);
      acc[tt] = wmma_bf16(A, B, acc[tt]);
    }
  }
  float* lb = logits + (size_t)b * NT * NP;
  int mofs = (lane < 16) ? 0 : 8;
#pragma unroll
  for (int tt = 0; tt < 8; ++tt)
#pragma unroll
    for (int r = 0; r < 8; ++r) {
      int t = tt*16 + r + mofs;
      lb[(size_t)t * NP + n0 + lo] = acc[tt][r];
    }
}

// ---------------------------------------------------------------------------
// K3b: Sinkhorn as alternating LSE on small vectors:
//   u[b,t] = LSE_n(logits - v[n]);  v[b,n] = LSE_t(logits - u[t])
// ---------------------------------------------------------------------------
__global__ void __launch_bounds__(256)
k_rowlse(const float* __restrict__ logits, const float* __restrict__ vs,
         float* __restrict__ u) {
  int b = blockIdx.x >> 7, t = blockIdx.x & 127;
  int tid = threadIdx.x;
  const float* row = logits + ((size_t)b * NT + t) * NP;
  const float* vb  = vs + b * NP;
  float m = -3.0e38f, s = 0.f;
  for (int n = tid; n < NP; n += 256) {
    float val = row[n] - vb[n];
    if (val > m) { s = s * __expf(m - val) + 1.f; m = val; }
    else         { s += __expf(val - m); }
  }
  __shared__ float mr[256], sr[256];
  mr[tid] = m; sr[tid] = s;
  __syncthreads();
  for (int off = 128; off > 0; off >>= 1) {
    if (tid < off) {
      float mA = mr[tid], sA = sr[tid], mB = mr[tid+off], sB = sr[tid+off];
      float mm = fmaxf(mA, mB);
      mr[tid] = mm;
      sr[tid] = sA * __expf(mA - mm) + sB * __expf(mB - mm);
    }
    __syncthreads();
  }
  if (tid == 0) u[b * NT + t] = mr[0] + __logf(sr[0]);
}

__global__ void __launch_bounds__(256)
k_collse(const float* __restrict__ logits, const float* __restrict__ u,
         float* __restrict__ vs) {
  int b = blockIdx.x >> 4, nb = blockIdx.x & 15;
  int tid = threadIdx.x;
  int n = nb * 256 + tid;
  __shared__ float ul[NT];
  if (tid < NT) ul[tid] = u[b * NT + tid];
  __syncthreads();
  const float* base = logits + (size_t)b * NT * NP + n;
  float m = -3.0e38f, s = 0.f;
  for (int t = 0; t < NT; ++t) {
    float val = base[(size_t)t * NP] - ul[t];
    if (val > m) { s = s * __expf(m - val) + 1.f; m = val; }
    else         { s += __expf(val - m); }
  }
  vs[b * NP + n] = m + __logf(s);
}

// ---------------------------------------------------------------------------
// K3c: tokens[b,t,d] = sum_n exp(logits-u-v) * v[b,n,d]  (fused exp, WMMA)
// Grid: b(8) x n-chunk(16, 256 n each).  Partials reduced by k_red16.
// ---------------------------------------------------------------------------
__global__ void __launch_bounds__(256)
k_tokens(const float* __restrict__ logits, const __bf16* __restrict__ vt,
         const float* __restrict__ u, const float* __restrict__ vs,
         float* __restrict__ partials) {
  int bx = blockIdx.x;
  int b = bx >> 4, c = bx & 15;
  int tid = threadIdx.x, w = tid >> 5, lane = tid & 31;
  int lo = lane & 15, dl = (lane < 16) ? 0 : 8;
  __shared__ float ul[NT];
  __shared__ float vl[256];
  if (tid < NT) ul[tid] = u[b * NT + tid];
  vl[tid] = vs[b * NP + c * 256 + tid];
  __syncthreads();

  int tt = w;                       // 8 waves = 8 t-tiles = all of T
  int trow = tt * 16 + lo;
  float ut = ul[trow];
  const float* lrow = logits + ((size_t)b * NT + trow) * NP + c * 256;
  v8f acc[16];
#pragma unroll
  for (int i = 0; i < 16; ++i) acc[i] = vzero8();

  for (int kt = 0; kt < 16; ++kt) {
    int nb = kt * 32 + dl;
    float f[16];
    ld8f(lrow + nb, f);
    ld8f(lrow + nb + 16, f + 8);
    BV16 At;
#pragma unroll
    for (int j = 0; j < 16; ++j) {
      int kk = nb + ((j < 8) ? j : j + 8);   // local n within this 256-chunk
      At.h[j] = (__bf16)__expf(f[j] - ut - vl[kk]);
    }
    v16bf A = At.v;
#pragma unroll
    for (int dt = 0; dt < 16; ++dt) {
      const __bf16* br =
          vt + (size_t)b * (ND*NP) + (size_t)(dt*16 + lo) * NP + c * 256;
      v16bf B = ld16bf(br + nb, br + nb + 16);
      acc[dt] = wmma_bf16(A, B, acc[dt]);
    }
  }
  float* pb = partials + (size_t)(c * NB + b) * NT * ND;
  int mofs = (lane < 16) ? 0 : 8;
#pragma unroll
  for (int dt = 0; dt < 16; ++dt)
#pragma unroll
    for (int r = 0; r < 8; ++r) {
      int t = tt*16 + r + mofs;
      pb[(size_t)t * ND + dt*16 + lo] = acc[dt][r];
    }
}

__global__ void __launch_bounds__(256)
k_red16(const float* __restrict__ p, float* __restrict__ t) {
  size_t i = (size_t)blockIdx.x * 256 + threadIdx.x;
  float s = 0.f;
#pragma unroll
  for (int c = 0; c < 16; ++c) s += p[(size_t)c * (NB*NT*ND) + i];
  t[i] = s;
}

// ---------------------------------------------------------------------------
// K4: out[b,n] = sum_{t,d} tokens[b,t,d] * out_proj[t,d,n]  (512 MB stream)
// Grid: kc(8 K-chunks) x nb(16 n-blocks).  Partials reduced by k_red8.
// ---------------------------------------------------------------------------
__global__ void __launch_bounds__(256)
k_outproj(const float* __restrict__ tokens, const float* __restrict__ op,
          float* __restrict__ outp) {
  int kc = blockIdx.x >> 4, nb = blockIdx.x & 15;
  int tid = threadIdx.x;
  int n = nb * 256 + tid;
  __shared__ __align__(16) float lt[256 * 8];
  float acc[8] = {0.f,0.f,0.f,0.f,0.f,0.f,0.f,0.f};
  for (int sub = 0; sub < 16; ++sub) {
    int r0 = kc * 4096 + sub * 256;
    __syncthreads();
#pragma unroll
    for (int b = 0; b < 8; ++b)
      lt[tid * 8 + b] = tokens[(size_t)b * (NT*ND) + r0 + tid];
    __syncthreads();
    for (int i = 0; i < 256; ++i) {
      float wv = op[(size_t)(r0 + i) * NP + n];
      const float4* tp = (const float4*)&lt[i * 8];
      float4 t0 = tp[0], t1 = tp[1];
      acc[0] += wv * t0.x; acc[1] += wv * t0.y;
      acc[2] += wv * t0.z; acc[3] += wv * t0.w;
      acc[4] += wv * t1.x; acc[5] += wv * t1.y;
      acc[6] += wv * t1.z; acc[7] += wv * t1.w;
    }
  }
#pragma unroll
  for (int b = 0; b < 8; ++b)
    outp[(size_t)(kc * NB + b) * NP + n] = acc[b];
}

__global__ void __launch_bounds__(256)
k_red8(const float* __restrict__ p, float* __restrict__ o) {
  size_t i = (size_t)blockIdx.x * 256 + threadIdx.x;
  float s = 0.f;
#pragma unroll
  for (int c = 0; c < 8; ++c) s += p[(size_t)c * (NB*NP) + i];
  o[i] = s;
}

// ---------------------------------------------------------------------------
extern "C" void kernel_launch(void* const* d_in, const int* in_sizes, int n_in,
                              void* d_out, int out_size, void* d_ws, size_t ws_size,
                              hipStream_t stream) {
  const float* x    = (const float*)d_in[0];   // [8,4096]
  const float* proj = (const float*)d_in[1];   // [4096,256,256]
  const float* qt   = (const float*)d_in[2];   // [128,256]
  const float* op   = (const float*)d_in[3];   // [128,256,4096]
  const float* Wq   = (const float*)d_in[4];   // [256,256]
  const float* Wk   = (const float*)d_in[5];
  const float* Wv   = (const float*)d_in[6];
  float* out = (float*)d_out;                  // [8,4096]

  char* ws = (char*)d_ws;
  const size_t MB = 1u << 20;
  __bf16* embeds = (__bf16*)(ws);                  // 16 MB
  __bf16* kbuf   = (__bf16*)(ws + 16*MB);          // 16 MB
  __bf16* vtbuf  = (__bf16*)(ws + 32*MB);          // 16 MB
  float* logits   = (float*)(ws);                  // reuse embeds
  float* partials = (float*)(ws + 16*MB);          // reuse kbuf
  char* sm = ws + 48*MB;
  __bf16* qb  = (__bf16*)(sm);                     // 64 KB
  __bf16* wkp = (__bf16*)(sm + 64*1024);           // 128 KB
  __bf16* wvp = (__bf16*)(sm + 192*1024);          // 128 KB
  float* ubuf  = (float*)(sm + 320*1024);          // 4 KB
  float* vsink = (float*)(sm + 328*1024);          // 128 KB
  float* tokens= (float*)(sm + 456*1024);          // 1 MB
  float* outp  = (float*)(sm + (456+1024)*1024);   // 1 MB

  k_qproj<<<NT, 256, 0, stream>>>(qt, Wq, qb);
  k_pack <<<32, 256, 0, stream>>>(Wk, Wv, wkp, wvp);
  k_zero <<<128, 256, 0, stream>>>(vsink);

  k_embed<<<NP, 256, 0, stream>>>(x, proj, embeds);
  k_kv   <<<256, 256, 0, stream>>>(embeds, wkp, kbuf, 0);
  k_kv   <<<256, 256, 0, stream>>>(embeds, wvp, vtbuf, 1);
  k_logits<<<256, 256, 0, stream>>>(qb, kbuf, logits);

  for (int it = 0; it < 5; ++it) {
    k_rowlse<<<NB*NT, 256, 0, stream>>>(logits, vsink, ubuf);
    k_collse<<<NB*16, 256, 0, stream>>>(logits, ubuf, vsink);
  }

  k_tokens<<<NB*16, 256, 0, stream>>>(logits, vtbuf, ubuf, vsink, partials);
  k_red16 <<<1024, 256, 0, stream>>>(partials, tokens);
  k_outproj<<<128, 256, 0, stream>>>(tokens, op, outp);
  k_red8  <<<128, 256, 0, stream>>>(outp, out);
}
